// RuleParse_84808424227020
// MI455X (gfx1250) — matrix operand order repaired and checked by reading, not compiled, not run on hardware
//
#include <hip/hip_runtime.h>
#include <hip/hip_bf16.h>
#include <math.h>

// ---------------------------------------------------------------------------
// CDNA5 WMMA types
// ---------------------------------------------------------------------------
typedef __attribute__((ext_vector_type(16))) __bf16 v16bf;
typedef __attribute__((ext_vector_type(8)))  __bf16 v8bf;
typedef __attribute__((ext_vector_type(8)))  float  v8f;

static __device__ __forceinline__ __bf16 tobf(float f) { return (__bf16)f; }

// ordered-uint encoding for float atomic max
static __device__ __forceinline__ unsigned encf(float f) {
  unsigned u = __float_as_uint(f);
  return (u & 0x80000000u) ? ~u : (u | 0x80000000u);
}
static __device__ __forceinline__ float decf(unsigned u) {
  u = (u & 0x80000000u) ? (u & 0x7FFFFFFFu) : ~u;
  return __uint_as_float(u);
}
#define ENC_NEG_INF 0x007FFFFFu  // encf(-inf)

// ---------------------------------------------------------------------------
// bf16 WMMA GEMM: C[M,Nc] = A[M,K] @ B[K,Nc] (+bias) (+relu)
// Block = 128 threads = 4 waves; block tile = 64(M) x 16(N); K stepped by 32.
// A tile (64x32) and transposed B tile (16x32, [n][k]) staged in LDS with
// branchless clamped loads (float4 fast path for interior tiles).
// Fragments per CDNA5 ISA 7.12.2: two contiguous 16B LDS reads each.
// ---------------------------------------------------------------------------
__global__ void gemm_wmma_bf16(const float* __restrict__ A,
                               const float* __restrict__ B,
                               const float* __restrict__ bias,
                               float* __restrict__ C,
                               int M, int K, int Nc, int fuse_relu) {
  __shared__ __align__(16) __bf16 sA[64 * 32];
  __shared__ __align__(16) __bf16 sB[16 * 32];   // transposed: [n][k]

  const int tid  = threadIdx.x;
  const int wave = tid >> 5;
  const int lane = tid & 31;
  const int m    = lane & 15;
  const int hi   = lane >> 4;
  const int tileM = blockIdx.y * 64;
  const int tileN = blockIdx.x * 16;

  // cooperative-load coordinates
  const int arow = tid >> 1;            // 0..63
  const int akk  = (tid & 1) * 16;      // 0 or 16
  const int brow = tid >> 2;            // 0..31
  const int bcq  = (tid & 3) * 4;       // 0,4,8,12

  const bool kvec = ((K & 3) == 0);
  const bool nvec = ((Nc & 3) == 0);

  v8f acc = {};
  for (int k0 = 0; k0 < K; k0 += 32) {
    const bool kfull = (k0 + 32 <= K);
    // ---- stage A tile (rows tileM..tileM+63, k0..k0+31) ----
    {
      const int  grow = tileM + arow;
      const bool rok  = grow < M;
      const float* ap = A + (size_t)(rok ? grow : (M - 1)) * K;
      __bf16* d = sA + arow * 32 + akk;
      if (rok && kfull && kvec) {
        const float4* s = (const float4*)(ap + k0 + akk);
#pragma unroll
        for (int j = 0; j < 4; ++j) {
          float4 f = s[j];
          d[4 * j + 0] = tobf(f.x); d[4 * j + 1] = tobf(f.y);
          d[4 * j + 2] = tobf(f.z); d[4 * j + 3] = tobf(f.w);
        }
      } else {
#pragma unroll
        for (int j = 0; j < 16; ++j) {
          const int k  = k0 + akk + j;
          const bool ok = rok && (k < K);
          float v = ap[ok ? k : 0];
          d[j] = tobf(ok ? v : 0.f);
        }
      }
    }
    // ---- stage B tile transposed (k0..k0+31 x tileN..tileN+15) ----
    {
      const int k = k0 + brow;
      if (kfull && nvec && (tileN + 16 <= Nc)) {
        float4 f = *(const float4*)(B + (size_t)k * Nc + tileN + bcq);
        sB[(bcq + 0) * 32 + brow] = tobf(f.x);
        sB[(bcq + 1) * 32 + brow] = tobf(f.y);
        sB[(bcq + 2) * 32 + brow] = tobf(f.z);
        sB[(bcq + 3) * 32 + brow] = tobf(f.w);
      } else {
        const bool kok = k < K;
        const float* bp = B + (size_t)(kok ? k : 0) * Nc;
#pragma unroll
        for (int j = 0; j < 4; ++j) {
          const int col = tileN + bcq + j;
          const bool ok = kok && (col < Nc);
          float v = bp[ok ? col : 0];
          sB[(bcq + j) * 32 + brow] = tobf(ok ? v : 0.f);
        }
      }
    }
    __syncthreads();

    // ---- build fragments from LDS (2 x 16B contiguous reads each) ----
    // A 16x32 bf16: lane(m,hi): k in {hi*8..hi*8+7} U {16+hi*8..16+hi*8+7}
    const v8bf* pa = (const v8bf*)(sA + (wave * 16 + m) * 32);
    union { v16bf v; v8bf h[2]; } ua;
    ua.h[0] = pa[hi];
    ua.h[1] = pa[hi + 2];
    // B 32x16 bf16: lane(n=m,hi): k in {hi*8..hi*8+7} U {16+hi*8..16+hi*8+7}
    const v8bf* pb = (const v8bf*)(sB + m * 32);
    union { v16bf v; v8bf h[2]; } ub;
    ub.h[0] = pb[hi];
    ub.h[1] = pb[hi + 2];

    acc = __builtin_amdgcn_wmma_f32_16x16x32_bf16(false, ua.v, false, ub.v,
                                                  (short)0, acc, false, false);
    __syncthreads();
  }

  // C 16x16 f32: VGPR r -> M = r + 8*hi, N = lane&15
  const int tm = tileM + wave * 16;
#pragma unroll
  for (int r = 0; r < 8; ++r) {
    const int row = tm + r + 8 * hi;
    const int col = tileN + m;
    if (row < M && col < Nc) {
      float v = acc[r];
      if (bias) v += bias[col];
      if (fuse_relu) v = fmaxf(v, 0.f);
      C[(size_t)row * Nc + col] = v;
    }
  }
}

// ---------------------------------------------------------------------------
// Fills
// ---------------------------------------------------------------------------
__global__ void fill_f32(float* p, float v, int n) {
  int i = blockIdx.x * blockDim.x + threadIdx.x;
  if (i < n) p[i] = v;
}
__global__ void fill_u32(unsigned* p, unsigned v, int n) {
  int i = blockIdx.x * blockDim.x + threadIdx.x;
  if (i < n) p[i] = v;
}

// ---------------------------------------------------------------------------
// Edge logits: one wave (32 lanes) per edge.
//   ea = edge_attr[e] @ We   (We cached in LDS, 18 x C)
//   s  = leaky_relu(xl[src]+xr[dst]+ea, 0.2)
//   logits[e] = dot(s, att);  atomicMax m_enc[dst]
// ---------------------------------------------------------------------------
__global__ void edge_logits(const float* __restrict__ xl,
                            const float* __restrict__ xr,
                            const float* __restrict__ eattr,
                            const int* __restrict__ src,
                            const int* __restrict__ dst,
                            const float* __restrict__ We,
                            const float* __restrict__ att,
                            float* __restrict__ logits,
                            unsigned* __restrict__ m_enc,
                            int E, int C) {
  __shared__ float sW[18 * 200];
  __shared__ float sAtt[200];
  for (int i = threadIdx.x; i < 18 * C; i += blockDim.x) sW[i] = We[i];
  for (int i = threadIdx.x; i < C; i += blockDim.x) sAtt[i] = att[i];
  __syncthreads();

  const int wave = threadIdx.x >> 5;
  const int lane = threadIdx.x & 31;
  const int e = blockIdx.x * (blockDim.x >> 5) + wave;
  if (e >= E) return;
  const int s = src[e], d = dst[e];

  float ea[18];
#pragma unroll
  for (int k = 0; k < 18; ++k) ea[k] = eattr[(size_t)e * 18 + k];

  float acc = 0.f;
  for (int c = lane; c < C; c += 32) {
    float v = 0.f;
#pragma unroll
    for (int k = 0; k < 18; ++k) v += ea[k] * sW[k * C + c];
    v += xl[(size_t)s * C + c] + xr[(size_t)d * C + c];
    v = (v > 0.f) ? v : 0.2f * v;
    acc += v * sAtt[c];
  }
#pragma unroll
  for (int off = 16; off; off >>= 1) acc += __shfl_xor(acc, off, 32);
  if (lane == 0) {
    logits[e] = acc;
    atomicMax(m_enc + d, encf(acc));
  }
}

// ---------------------------------------------------------------------------
// ex[e] = exp(logit - m[dst]);  denom[dst] += ex
// ---------------------------------------------------------------------------
__global__ void edge_exp(const float* __restrict__ logits,
                         const int* __restrict__ dst,
                         const unsigned* __restrict__ m_enc,
                         float* __restrict__ ex,
                         float* __restrict__ denom, int E) {
  int e = blockIdx.x * blockDim.x + threadIdx.x;
  if (e >= E) return;
  int d = dst[e];
  float v = __expf(logits[e] - decf(m_enc[d]));
  ex[e] = v;
  atomicAdd(denom + d, v);
}

// ---------------------------------------------------------------------------
// out[dst] += (ex/denom[dst]) * xl[src] — one wave per edge
// ---------------------------------------------------------------------------
__global__ void edge_scatter(const float* __restrict__ xl,
                             const int* __restrict__ src,
                             const int* __restrict__ dst,
                             const float* __restrict__ ex,
                             const float* __restrict__ denom,
                             float* __restrict__ out, int E, int C) {
  const int wave = threadIdx.x >> 5;
  const int lane = threadIdx.x & 31;
  const int e = blockIdx.x * (blockDim.x >> 5) + wave;
  if (e >= E) return;
  const int s = src[e], d = dst[e];
  const float coef = ex[e] / fmaxf(denom[d], 1e-16f);
  for (int c = lane; c < C; c += 32)
    atomicAdd(out + (size_t)d * C + c, coef * xl[(size_t)s * C + c]);
}

// ---------------------------------------------------------------------------
// out = in + bias  (+relu)
// ---------------------------------------------------------------------------
__global__ void bias_act(const float* __restrict__ in,
                         const float* __restrict__ bias,
                         float* __restrict__ out, int M, int C, int relu) {
  size_t i = (size_t)blockIdx.x * blockDim.x + threadIdx.x;
  if (i >= (size_t)M * C) return;
  int c = (int)(i % C);
  float v = in[i] + bias[c];
  if (relu) v = fmaxf(v, 0.f);
  out[i] = v;
}

// ---------------------------------------------------------------------------
// BatchNorm (training-mode batch stats, biased variance)
// Row-major coalesced partial sums; each thread owns <=2 channels in regs.
// ---------------------------------------------------------------------------
__global__ void bn_partial(const float* __restrict__ h, int M, int C,
                           float* __restrict__ sum, float* __restrict__ sumsq) {
  const int c0 = threadIdx.x;
  const int c1 = c0 + 256;
  float s0 = 0.f, q0 = 0.f, s1 = 0.f, q1 = 0.f;
  for (int r = blockIdx.x; r < M; r += gridDim.x) {
    const float* row = h + (size_t)r * C;
    if (c0 < C) { float v = row[c0]; s0 += v; q0 += v * v; }
    if (c1 < C) { float v = row[c1]; s1 += v; q1 += v * v; }
  }
  if (c0 < C) { atomicAdd(&sum[c0], s0); atomicAdd(&sumsq[c0], q0); }
  if (c1 < C) { atomicAdd(&sum[c1], s1); atomicAdd(&sumsq[c1], q1); }
}

__global__ void bn_finalize(const float* __restrict__ sum,
                            const float* __restrict__ sumsq, int M, int C,
                            float* __restrict__ mean, float* __restrict__ istd) {
  int c = blockIdx.x * blockDim.x + threadIdx.x;
  if (c >= C) return;
  float mu = sum[c] / M;
  float var = sumsq[c] / M - mu * mu;
  mean[c] = mu;
  istd[c] = rsqrtf(var + 1e-5f);
}

__global__ void bn_apply(float* __restrict__ h, int M, int C,
                         const float* __restrict__ mean,
                         const float* __restrict__ istd,
                         const float* __restrict__ g,
                         const float* __restrict__ b) {
  size_t i = (size_t)blockIdx.x * blockDim.x + threadIdx.x;
  if (i >= (size_t)M * C) return;
  int c = (int)(i % C);
  h[i] = (h[i] - mean[c]) * istd[c] * g[c] + b[c];
}

// ---------------------------------------------------------------------------
// Global mean pool: one wave per node
// ---------------------------------------------------------------------------
__global__ void pool_sum(const float* __restrict__ h,
                         const int* __restrict__ bids,
                         float* __restrict__ sums,
                         float* __restrict__ cnt, int Nn, int C) {
  const int wave = threadIdx.x >> 5;
  const int lane = threadIdx.x & 31;
  const int i = blockIdx.x * (blockDim.x >> 5) + wave;
  if (i >= Nn) return;
  const int b = bids[i];
  for (int c = lane; c < C; c += 32)
    atomicAdd(sums + (size_t)b * C + c, h[(size_t)i * C + c]);
  if (lane == 0) atomicAdd(cnt + b, 1.0f);
}

__global__ void pool_div(const float* __restrict__ sums,
                         const float* __restrict__ cnt,
                         float* __restrict__ p, int G, int C) {
  int i = blockIdx.x * blockDim.x + threadIdx.x;
  if (i >= G * C) return;
  p[i] = sums[i] / fmaxf(cnt[i / C], 1.f);
}

// ---------------------------------------------------------------------------
// Host-side orchestration
// ---------------------------------------------------------------------------
static inline int cdiv(long long a, long long b) { return (int)((a + b - 1) / b); }

extern "C" void kernel_launch(void* const* d_in, const int* in_sizes, int n_in,
                              void* d_out, int out_size, void* d_ws, size_t ws_size,
                              hipStream_t stream) {
  const int N = in_sizes[0] / 16;   // 50000
  const int E = in_sizes[1] / 2;    // 800000
  const int G = 256;
  const int C1 = 100, C2 = 200, C3 = 400, F1 = 200, F2 = 100;

  const float* x      = (const float*)d_in[0];
  const int*   ei     = (const int*)  d_in[1];
  const float* eattr  = (const float*)d_in[2];
  const int*   bids   = (const int*)  d_in[3];
  const float* Wl1 = (const float*)d_in[4];  const float* Wr1 = (const float*)d_in[5];
  const float* We1 = (const float*)d_in[6];  const float* att1= (const float*)d_in[7];
  const float* bias1=(const float*)d_in[8];
  const float* Wl2 = (const float*)d_in[9];  const float* Wr2 = (const float*)d_in[10];
  const float* We2 = (const float*)d_in[11]; const float* att2= (const float*)d_in[12];
  const float* bias2=(const float*)d_in[13];
  const float* W3  = (const float*)d_in[14]; const float* b3  = (const float*)d_in[15];
  const float* bn1g=(const float*)d_in[16];  const float* bn1b=(const float*)d_in[17];
  const float* bn2g=(const float*)d_in[18];  const float* bn2b=(const float*)d_in[19];
  const float* bn3g=(const float*)d_in[20];  const float* bn3b=(const float*)d_in[21];
  const float* bnpg=(const float*)d_in[22];  const float* bnpb=(const float*)d_in[23];
  const float* Wf1 = (const float*)d_in[24]; const float* bf1 = (const float*)d_in[25];
  const float* Wf2 = (const float*)d_in[26]; const float* bf2 = (const float*)d_in[27];
  const float* bnfg=(const float*)d_in[28];  const float* bnfb=(const float*)d_in[29];
  const float* Wf3 = (const float*)d_in[30]; const float* bf3 = (const float*)d_in[31];

  const int* src = ei;
  const int* dst = ei + E;

  // workspace carve-out (256B aligned)
  char* wsp = (char*)d_ws;
  auto alloc = [&](size_t elems) {
    void* p = (void*)wsp;
    wsp += ((elems * 4 + 255) / 256) * 256;
    return p;
  };
  float*    xl     = (float*)   alloc((size_t)N * C2);
  float*    xr     = (float*)   alloc((size_t)N * C2);
  float*    agg    = (float*)   alloc((size_t)N * C2);
  float*    h      = (float*)   alloc((size_t)N * C2);
  float*    h3     = (float*)   alloc((size_t)N * C3);
  float*    logits = (float*)   alloc(E);
  float*    ex     = (float*)   alloc(E);
  unsigned* m_enc  = (unsigned*)alloc(N);
  float*    denom  = (float*)   alloc(N);
  float*    sums   = (float*)   alloc((size_t)G * C3);
  float*    cnt    = (float*)   alloc(G);
  float*    p      = (float*)   alloc((size_t)G * C3);
  float*    q1     = (float*)   alloc((size_t)G * F1);
  float*    q2     = (float*)   alloc((size_t)G * F2);
  float*    mean   = (float*)   alloc(C3);
  float*    istd   = (float*)   alloc(C3);
  float*    accS   = (float*)   alloc(C3);
  float*    accQ   = (float*)   alloc(C3);

  const int EW = cdiv(E, 8);   // wave-per-edge blocks (8 waves / 256 thr)

  auto gemm = [&](const float* A, const float* B, const float* bias, float* C,
                  int M, int K, int Nc, int relu) {
    dim3 grid(cdiv(Nc, 16), cdiv(M, 64));
    gemm_wmma_bf16<<<grid, 128, 0, stream>>>(A, B, bias, C, M, K, Nc, relu);
  };
  auto bn = [&](float* buf, int M, int C, const float* g, const float* b) {
    fill_f32<<<cdiv(C, 256), 256, 0, stream>>>(accS, 0.f, C);
    fill_f32<<<cdiv(C, 256), 256, 0, stream>>>(accQ, 0.f, C);
    int blocks = M < 768 ? M : 768;
    bn_partial<<<blocks, 256, 0, stream>>>(buf, M, C, accS, accQ);
    bn_finalize<<<cdiv(C, 256), 256, 0, stream>>>(accS, accQ, M, C, mean, istd);
    bn_apply<<<cdiv((long long)M * C, 256), 256, 0, stream>>>(buf, M, C, mean, istd, g, b);
  };
  auto gat = [&](const float* We, const float* att, const float* bias,
                 const float* g, const float* b, int C) {
    fill_u32<<<cdiv(N, 256), 256, 0, stream>>>(m_enc, ENC_NEG_INF, N);
    fill_f32<<<cdiv(N, 256), 256, 0, stream>>>(denom, 0.f, N);
    fill_f32<<<cdiv((long long)N * C, 256), 256, 0, stream>>>(agg, 0.f, N * C);
    edge_logits<<<EW, 256, 0, stream>>>(xl, xr, eattr, src, dst, We, att,
                                        logits, m_enc, E, C);
    edge_exp<<<cdiv(E, 256), 256, 0, stream>>>(logits, dst, m_enc, ex, denom, E);
    edge_scatter<<<EW, 256, 0, stream>>>(xl, src, dst, ex, denom, agg, E, C);
    bias_act<<<cdiv((long long)N * C, 256), 256, 0, stream>>>(agg, bias, h, N, C, 1);
    bn(h, N, C, g, b);
  };

  // ---- layer 1: 16 -> 100 ----
  gemm(x, Wl1, nullptr, xl, N, 16, C1, 0);
  gemm(x, Wr1, nullptr, xr, N, 16, C1, 0);
  gat(We1, att1, bias1, bn1g, bn1b, C1);

  // ---- layer 2: 100 -> 200 ----
  gemm(h, Wl2, nullptr, xl, N, C1, C2, 0);
  gemm(h, Wr2, nullptr, xr, N, C1, C2, 0);
  gat(We2, att2, bias2, bn2g, bn2b, C2);

  // ---- linear 200 -> 400 + BN ----
  gemm(h, W3, b3, h3, N, C2, C3, 0);
  bn(h3, N, C3, bn3g, bn3b);

  // ---- global mean pool + BN ----
  fill_f32<<<cdiv(G * C3, 256), 256, 0, stream>>>(sums, 0.f, G * C3);
  fill_f32<<<1, 256, 0, stream>>>(cnt, 0.f, G);
  pool_sum<<<cdiv(N, 8), 256, 0, stream>>>(h3, bids, sums, cnt, N, C3);
  pool_div<<<cdiv(G * C3, 256), 256, 0, stream>>>(sums, cnt, p, G, C3);
  bn(p, G, C3, bnpg, bnpb);

  // ---- FFN 400 -> 200 -> 100 -> BN -> 100 ----
  gemm(p, Wf1, bf1, q1, G, C3, F1, 1);
  gemm(q1, Wf2, bf2, q2, G, F1, F2, 1);
  bn(q2, G, F2, bnfg, bnfb);
  gemm(q2, Wf3, bf3, (float*)d_out, G, F2, F2, 0);
}
